// clusteringLayer_77154792506116
// MI455X (gfx1250) — compile-verified, matrix operand order checked by reading, not run on hardware
//
#include <hip/hip_runtime.h>
#include <hip/hip_bf16.h>

typedef __attribute__((ext_vector_type(16))) _Float16 v16h;
typedef __attribute__((ext_vector_type(8)))  float    v8f;

#define BB 2048
#define FF 256   // in_features == n_clusters == 256

// ---------------------------------------------------------------------------
// Kernel 1: transpose mu (C x F, row-major mu[j][i]) -> muT (F x C, muT[i][j])
// Tiny (256 KB) one-shot; LDS tile keeps both sides coalesced.
// ---------------------------------------------------------------------------
__global__ __launch_bounds__(256) void mu_transpose_kernel(
    const float* __restrict__ mu, float* __restrict__ muT) {
  __shared__ float tile[16][17];
  const int tx = threadIdx.x, ty = threadIdx.y;
  const int j0 = blockIdx.y * 16, i0 = blockIdx.x * 16;
  tile[ty][tx] = mu[(j0 + ty) * FF + (i0 + tx)];          // coalesced read
  __syncthreads();
  muT[(i0 + ty) * FF + (j0 + tx)] = tile[tx][ty];          // coalesced write
}

// ---------------------------------------------------------------------------
// Kernel 2: one block per batch row b (256 threads = 8 waves, wave32).
//   Reference: t = sqrt(2)*rsqrt(1+d^2), out = num/den.  The sqrt(2) factor
//   cancels exactly in num/den, so we sum t' = rsqrt(1+d^2) instead — one
//   fewer VALU op per element in the compute-bound inner loop.
//   num[i] = sum_j t'  via v_wmma_f32_16x16x32_f16 with all-ones B matrix
//   den    = sum_i num[i] (LDS tree reduction); out = num/den
// A-matrix layout (16-bit, 16x32): lanes l and l+16 share row M=l and
// together cover all 32 K-slots; since B == 1 for every (k,n), any j->K
// permutation yields the exact row sum, so lane l takes j=jb..jb+15 and
// lane l+16 takes j=jb+16..jb+31.
// ---------------------------------------------------------------------------
__global__ __launch_bounds__(256) void cluster_soft_assign_kernel(
    const float* __restrict__ x, const float* __restrict__ muT,
    float* __restrict__ out) {
  const int b    = blockIdx.x;
  const int tid  = threadIdx.x;
  const int lane = tid & 31;
  const int wave = tid >> 5;
  const int m    = lane & 15;       // row within i-block
  const int half = lane >> 4;       // 0: j low 16, 1: j high 16

  __shared__ float xs[FF];          // x row, broadcast-read
  __shared__ float nums[FF];        // per-cluster numerators
  __shared__ float red[256];        // reduction scratch

  xs[tid] = x[b * FF + tid];
  __syncthreads();

  v16h ones;
#pragma unroll
  for (int k = 0; k < 16; ++k) ones[k] = (_Float16)1.0f;

#pragma unroll
  for (int blk = 0; blk < 2; ++blk) {
    const int ib = wave * 2 + blk;            // i-block 0..15
    const int i  = ib * 16 + m;               // this lane's output row
    const float* __restrict__ mrow = muT + i * FF;

    v8f c = {0.f, 0.f, 0.f, 0.f, 0.f, 0.f, 0.f, 0.f};

#pragma unroll
    for (int jb = 0; jb < FF; jb += 32) {
      const int joff = jb + half * 16;
      v16h a;
#pragma unroll
      for (int k = 0; k < 16; ++k) {
        const float d = xs[joff + k] - mrow[joff + k];   // b128 row stream
        const float s = __builtin_fmaf(d, d, 1.0f);      // 1 + d^2
        a[k] = (_Float16)__builtin_amdgcn_rsqf(s);       // t' (sqrt2 cancels)
      }
      // D = A(16x32 f16) * Ones(32x16) + C : every column n of D holds the
      // K-sum of row M, i.e. num[i]. 8 iterations -> full K=256 reduction.
      c = __builtin_amdgcn_wmma_f32_16x16x32_f16(
          /*neg_a=*/false, a, /*neg_b=*/false, ones,
          /*c_mod=*/(short)0, c, /*reuse_a=*/false, /*reuse_b=*/false);
    }

    // C/D layout: VGPR r, lanes 0-15 -> M=r col=lane; lanes 16-31 -> M=8+r.
    // Columns are identical, so lane 0 / lane 16 each spill 8 rows.
    if (lane == 0) {
#pragma unroll
      for (int r = 0; r < 8; ++r) nums[ib * 16 + r] = c[r];
    } else if (lane == 16) {
#pragma unroll
      for (int r = 0; r < 8; ++r) nums[ib * 16 + 8 + r] = c[r];
    }
  }
  __syncthreads();

  const float mynum = nums[tid];
  red[tid] = mynum;
  __syncthreads();
#pragma unroll
  for (int s = 128; s > 0; s >>= 1) {
    if (tid < s) red[tid] += red[tid + s];
    __syncthreads();
  }
  out[b * FF + tid] = mynum / red[0];
}

// ---------------------------------------------------------------------------
extern "C" void kernel_launch(void* const* d_in, const int* in_sizes, int n_in,
                              void* d_out, int out_size, void* d_ws, size_t ws_size,
                              hipStream_t stream) {
  const float* x  = (const float*)d_in[0];   // (B, F) fp32
  const float* mu = (const float*)d_in[1];   // (C, F) fp32
  float* out = (float*)d_out;                // (B, F) fp32
  float* muT = (float*)d_ws;                 // (F, C) fp32, 256 KB scratch

  dim3 tgrid(FF / 16, FF / 16), tblk(16, 16);
  mu_transpose_kernel<<<tgrid, tblk, 0, stream>>>(mu, muT);

  cluster_soft_assign_kernel<<<BB, 256, 0, stream>>>(x, muT, out);
}
